// PNA_19404662243713
// MI455X (gfx1250) — matrix operand order repaired and checked by reading, not compiled, not run on hardware
//
#include <hip/hip_runtime.h>

#define NNODES 50000
#define NEDGES 500000
#define F 128
#define CLS 40

typedef __attribute__((ext_vector_type(16))) __bf16 v16bf;
typedef __attribute__((ext_vector_type(8)))  float  v8f;

union Frag { unsigned u[8]; uint4 q[2]; v16bf v; };

__device__ __forceinline__ unsigned f2bf(float f) {
    unsigned u = __float_as_uint(f);
    return (u + 0x7FFFu + ((u >> 16) & 1u)) >> 16;   // round-to-nearest-even
}
__device__ __forceinline__ unsigned pk2(float a, float b) {
    return f2bf(a) | (f2bf(b) << 16);
}
__device__ __forceinline__ void loadB(Frag& f, const unsigned* bBase, int idx) {
    const unsigned* bp = bBase + idx * 256;
    f.q[0] = *(const uint4*)(bp);
    f.q[1] = *(const uint4*)(bp + 4);
}
__device__ __forceinline__ void loadA(Frag& f, const unsigned* arow) {
    f.q[0] = *(const uint4*)(arow);
    f.q[1] = *(const uint4*)(arow + 8);
}

// ---------------- conversion / init kernels ----------------

__global__ void k_conv_x(const float* __restrict__ x, unsigned* __restrict__ xb, int npairs) {
    int i = blockIdx.x * blockDim.x + threadIdx.x;
    if (i >= npairs) return;
    float2 v = ((const float2*)x)[i];
    xb[i] = pk2(v.x, v.y);
}

// W[K][Nn] fp32 -> fragment-major bf16 u32:
//   idx = ((kk*NT + nt)*2 + hs)*128 + col*8 + v
//   holds pair row p = kk*16 + (v>>2)*8 + hs*4 + (v&3), column n = nt*16 + col (zero-pad n >= Nn)
__global__ void k_conv_w(const float* __restrict__ w, unsigned* __restrict__ wb,
                         int K, int Nn, int NT) {
    int i = blockIdx.x * blockDim.x + threadIdx.x;
    int total = (K / 32) * NT * 256;
    if (i >= total) return;
    int v    = i & 7;
    int col  = (i >> 3) & 15;
    int hs   = (i >> 7) & 1;
    int rest = i >> 8;
    int nt   = rest % NT;
    int kk   = rest / NT;
    int p = kk * 16 + (v >> 2) * 8 + hs * 4 + (v & 3);
    int n = nt * 16 + col;
    unsigned r = 0;
    if (n < Nn) r = pk2(w[(2 * p) * Nn + n], w[(2 * p + 1) * Nn + n]);
    wb[i] = r;
}

__global__ void k_init_aggr(float* __restrict__ s, float* __restrict__ m, int total) {
    int i = blockIdx.x * blockDim.x + threadIdx.x;
    if (i >= total) return;
    s[i] = 0.0f;
    m[i] = -3.0e38f;
}

__global__ void k_init_cnt(float* __restrict__ c, int n) {
    int i = blockIdx.x * blockDim.x + threadIdx.x;
    if (i < n) c[i] = 0.0f;
}

__global__ void k_count(const int* __restrict__ ei, float* __restrict__ cnt) {
    int e = blockIdx.x * blockDim.x + threadIdx.x;
    if (e >= NEDGES) return;
    atomicAdd(&cnt[ei[NEDGES + e]], 1.0f);
}

// ---------------- edge message GEMM + scatter aggregate ----------------
// per wave: 16 edges.  A[16x256] = concat(x[dst], x[src]); double-buffered A/B
// fragments so fragment i+1 is in flight while WMMA consumes fragment i.
__global__ void k_edge(const unsigned* __restrict__ xb, const unsigned* __restrict__ wB,
                       const float* __restrict__ bias, const int* __restrict__ ei,
                       float* __restrict__ sum, float* __restrict__ mx) {
    int gid  = blockIdx.x * blockDim.x + threadIdx.x;
    int wave = gid >> 5, lane = gid & 31;
    const int tiles = NEDGES / 16;
    if (wave >= tiles) return;
    int row = lane & 15, hs = lane >> 4;
    int e    = wave * 16 + row;
    int srcn = ei[e];
    int dstn = ei[NEDGES + e];

    const unsigned* aDst  = xb + (size_t)dstn * 64 + hs * 4;
    const unsigned* aSrc  = xb + (size_t)srcn * 64 + hs * 4;
    const unsigned* bBase = wB + hs * 128 + row * 8;

    Frag a[2], b[2];
    loadA(a[0], aDst);
    loadB(b[0], bBase, 0);

    v8f acc[8] = {};
    #pragma unroll
    for (int kk = 0; kk < 8; ++kk) {
        if (kk + 1 < 8) {
            const unsigned* an = ((kk + 1) < 4 ? aDst : aSrc) + ((kk + 1) & 3) * 16;
            loadA(a[(kk + 1) & 1], an);
        }
        #pragma unroll
        for (int nt = 0; nt < 8; ++nt) {
            int li = kk * 8 + nt;
            if (li + 1 < 64) loadB(b[(li + 1) & 1], bBase, li + 1);
            acc[nt] = __builtin_amdgcn_wmma_f32_16x16x32_bf16(
                false, a[kk & 1].v, false, b[li & 1].v, (short)0, acc[nt], false, false);
        }
    }

    int dstm[8];
    #pragma unroll
    for (int r = 0; r < 8; ++r) dstm[r] = __shfl(dstn, r + hs * 8, 32);

    #pragma unroll
    for (int nt = 0; nt < 8; ++nt) {
        int n = nt * 16 + row;
        float bv = bias[n];
        #pragma unroll
        for (int r = 0; r < 8; ++r) {
            float val = acc[nt][r] + bv;
            size_t idx = (size_t)dstm[r] * F + n;
            atomicAdd(&sum[idx], val);
            unsafeAtomicMax(&mx[idx], val);   // global_atomic_max_num_f32
        }
    }
}

// ---------------- node post GEMM: [h | mean | max | sum] @ w_post + b ----------------
__global__ void k_post(const unsigned* __restrict__ hb, const float* __restrict__ sum,
                       const float* __restrict__ mx, const float* __restrict__ cnt,
                       const unsigned* __restrict__ wB, const float* __restrict__ bias,
                       unsigned short* __restrict__ tb) {
    int gid  = blockIdx.x * blockDim.x + threadIdx.x;
    int wave = gid >> 5, lane = gid & 31;
    const int tiles = NNODES / 16;
    if (wave >= tiles) return;
    int row = lane & 15, hs = lane >> 4;
    int node = wave * 16 + row;
    float c   = cnt[node];
    float inv = 1.0f / fmaxf(c, 1.0f);
    float msk = (c > 0.0f) ? 1.0f : 0.0f;

    const unsigned* aBase = hb + (size_t)node * 64 + hs * 4;
    const float*    sBase = sum + (size_t)node * F + hs * 8;
    const float*    mBase = mx  + (size_t)node * F + hs * 8;
    const unsigned* bBase = wB + hs * 128 + row * 8;

    // builds the A fragment for K-step kk (sections: 0=h 1=mean 2=max 3=sum)
    auto makeA = [&](Frag& a, int kk) {
        int sec = kk >> 2;
        if (sec == 0) {
            loadA(a, aBase + (kk & 3) * 16);
        } else {
            const float* fp = ((sec == 2) ? mBase : sBase) + (kk & 3) * 32;
            float4 fa = *(const float4*)(fp);
            float4 fb = *(const float4*)(fp + 4);
            float4 fc = *(const float4*)(fp + 16);
            float4 fd = *(const float4*)(fp + 20);
            float s = (sec == 1) ? inv : (sec == 2) ? msk : 1.0f;
            a.u[0] = pk2(fa.x * s, fa.y * s);
            a.u[1] = pk2(fa.z * s, fa.w * s);
            a.u[2] = pk2(fb.x * s, fb.y * s);
            a.u[3] = pk2(fb.z * s, fb.w * s);
            a.u[4] = pk2(fc.x * s, fc.y * s);
            a.u[5] = pk2(fc.z * s, fc.w * s);
            a.u[6] = pk2(fd.x * s, fd.y * s);
            a.u[7] = pk2(fd.z * s, fd.w * s);
        }
    };

    Frag a[2], b[2];
    makeA(a[0], 0);
    loadB(b[0], bBase, 0);

    v8f acc[8] = {};
    #pragma unroll
    for (int kk = 0; kk < 16; ++kk) {
        if (kk + 1 < 16) makeA(a[(kk + 1) & 1], kk + 1);
        #pragma unroll
        for (int nt = 0; nt < 8; ++nt) {
            int li = kk * 8 + nt;
            if (li + 1 < 128) loadB(b[(li + 1) & 1], bBase, li + 1);
            acc[nt] = __builtin_amdgcn_wmma_f32_16x16x32_bf16(
                false, a[kk & 1].v, false, b[li & 1].v, (short)0, acc[nt], false, false);
        }
    }

    #pragma unroll
    for (int nt = 0; nt < 8; ++nt) {
        int n = nt * 16 + row;
        float bv = bias[n];
        #pragma unroll
        for (int r = 0; r < 8; ++r) {
            int nodem = wave * 16 + r + hs * 8;
            tb[(size_t)nodem * F + n] = (unsigned short)f2bf(acc[nt][r] + bv);
        }
    }
}

// ---------------- 128x128 GEMM + bias + ReLU -> bf16 ----------------
__global__ void k_lin(const unsigned* __restrict__ ab, const unsigned* __restrict__ wB,
                      const float* __restrict__ bias, unsigned short* __restrict__ out) {
    int gid  = blockIdx.x * blockDim.x + threadIdx.x;
    int wave = gid >> 5, lane = gid & 31;
    const int tiles = NNODES / 16;
    if (wave >= tiles) return;
    int row = lane & 15, hs = lane >> 4;
    int node = wave * 16 + row;

    const unsigned* aBase = ab + (size_t)node * 64 + hs * 4;
    const unsigned* bBase = wB + hs * 128 + row * 8;

    Frag a[2], b[2];
    loadA(a[0], aBase);
    loadB(b[0], bBase, 0);

    v8f acc[8] = {};
    #pragma unroll
    for (int kk = 0; kk < 4; ++kk) {
        if (kk + 1 < 4) loadA(a[(kk + 1) & 1], aBase + (kk + 1) * 16);
        #pragma unroll
        for (int nt = 0; nt < 8; ++nt) {
            int li = kk * 8 + nt;
            if (li + 1 < 32) loadB(b[(li + 1) & 1], bBase, li + 1);
            acc[nt] = __builtin_amdgcn_wmma_f32_16x16x32_bf16(
                false, a[kk & 1].v, false, b[li & 1].v, (short)0, acc[nt], false, false);
        }
    }

    #pragma unroll
    for (int nt = 0; nt < 8; ++nt) {
        int n = nt * 16 + row;
        float bv = bias[n];
        #pragma unroll
        for (int r = 0; r < 8; ++r) {
            int nodem = wave * 16 + r + hs * 8;
            float val = fmaxf(acc[nt][r] + bv, 0.0f);      // ReLU
            out[(size_t)nodem * F + n] = (unsigned short)f2bf(val);
        }
    }
}

// ---------------- final classifier: [N,128] @ [128,40] + b -> f32 ----------------
__global__ void k_out(const unsigned* __restrict__ ab, const unsigned* __restrict__ wB,
                      const float* __restrict__ bias, float* __restrict__ out) {
    int gid  = blockIdx.x * blockDim.x + threadIdx.x;
    int wave = gid >> 5, lane = gid & 31;
    const int tiles = NNODES / 16;
    if (wave >= tiles) return;
    int row = lane & 15, hs = lane >> 4;
    int node = wave * 16 + row;

    const unsigned* aBase = ab + (size_t)node * 64 + hs * 4;
    const unsigned* bBase = wB + hs * 128 + row * 8;

    Frag a[2], b[2];
    loadA(a[0], aBase);
    loadB(b[0], bBase, 0);

    v8f acc[3] = {};
    #pragma unroll
    for (int kk = 0; kk < 4; ++kk) {
        if (kk + 1 < 4) loadA(a[(kk + 1) & 1], aBase + (kk + 1) * 16);
        #pragma unroll
        for (int nt = 0; nt < 3; ++nt) {
            int li = kk * 3 + nt;
            if (li + 1 < 12) loadB(b[(li + 1) & 1], bBase, li + 1);
            acc[nt] = __builtin_amdgcn_wmma_f32_16x16x32_bf16(
                false, a[kk & 1].v, false, b[li & 1].v, (short)0, acc[nt], false, false);
        }
    }

    #pragma unroll
    for (int nt = 0; nt < 3; ++nt) {
        int n = nt * 16 + row;
        if (n >= CLS) continue;
        float bv = bias[n];
        #pragma unroll
        for (int r = 0; r < 8; ++r) {
            int nodem = wave * 16 + r + hs * 8;
            out[(size_t)nodem * CLS + n] = acc[nt][r] + bv;
        }
    }
}

// ---------------- host launcher ----------------

extern "C" void kernel_launch(void* const* d_in, const int* in_sizes, int n_in,
                              void* d_out, int out_size, void* d_ws, size_t ws_size,
                              hipStream_t stream) {
    const float* x      = (const float*)d_in[0];
    const int*   ei     = (const int*)d_in[1];
    const float* w1_pre = (const float*)d_in[2];  const float* b1_pre = (const float*)d_in[3];
    const float* w1_post= (const float*)d_in[4];  const float* b1_post= (const float*)d_in[5];
    const float* w1_lin = (const float*)d_in[6];  const float* b1_lin = (const float*)d_in[7];
    const float* w2_pre = (const float*)d_in[8];  const float* b2_pre = (const float*)d_in[9];
    const float* w2_post= (const float*)d_in[10]; const float* b2_post= (const float*)d_in[11];
    const float* w2_lin = (const float*)d_in[12]; const float* b2_lin = (const float*)d_in[13];
    const float* w_out  = (const float*)d_in[14]; const float* b_out  = (const float*)d_in[15];

    char* base = (char*)d_ws;
    size_t off = 0;
    auto alloc = [&](size_t bytes) -> char* {
        char* p = base + off;
        off += (bytes + 255) & ~(size_t)255;
        return p;
    };
    unsigned* xb   = (unsigned*)alloc((size_t)NNODES * 64 * 4);   // x  bf16 pairs
    unsigned* tb   = (unsigned*)alloc((size_t)NNODES * 64 * 4);   // t  bf16 pairs (temp)
    unsigned* h1b  = (unsigned*)alloc((size_t)NNODES * 64 * 4);   // h1 bf16 pairs
    unsigned* h2b  = (unsigned*)alloc((size_t)NNODES * 64 * 4);   // h2 bf16 pairs
    float*    sum  = (float*)alloc((size_t)NNODES * F * 4);
    float*    mx   = (float*)alloc((size_t)NNODES * F * 4);
    float*    cnt  = (float*)alloc((size_t)NNODES * 4);
    unsigned* wpre1  = (unsigned*)alloc(8  * 8 * 256 * 4);   // K=256, NT=8
    unsigned* wpost1 = (unsigned*)alloc(16 * 8 * 256 * 4);   // K=512, NT=8
    unsigned* wlin1  = (unsigned*)alloc(4  * 8 * 256 * 4);   // K=128, NT=8
    unsigned* wpre2  = (unsigned*)alloc(8  * 8 * 256 * 4);
    unsigned* wpost2 = (unsigned*)alloc(16 * 8 * 256 * 4);
    unsigned* wlin2  = (unsigned*)alloc(4  * 8 * 256 * 4);
    unsigned* wob    = (unsigned*)alloc(4  * 3 * 256 * 4);   // K=128, NT=3 (CLS 40 -> 48)

    const int BT = 256;
    auto nb = [](long total, int bt) { return (int)((total + bt - 1) / bt); };

    // conversions
    k_conv_x<<<nb((long)NNODES * 64, BT), BT, 0, stream>>>(x, xb, NNODES * 64);
    k_conv_w<<<nb(8  * 8 * 256, BT), BT, 0, stream>>>(w1_pre,  wpre1,  256, 128, 8);
    k_conv_w<<<nb(16 * 8 * 256, BT), BT, 0, stream>>>(w1_post, wpost1, 512, 128, 8);
    k_conv_w<<<nb(4  * 8 * 256, BT), BT, 0, stream>>>(w1_lin,  wlin1,  128, 128, 8);
    k_conv_w<<<nb(8  * 8 * 256, BT), BT, 0, stream>>>(w2_pre,  wpre2,  256, 128, 8);
    k_conv_w<<<nb(16 * 8 * 256, BT), BT, 0, stream>>>(w2_post, wpost2, 512, 128, 8);
    k_conv_w<<<nb(4  * 8 * 256, BT), BT, 0, stream>>>(w2_lin,  wlin2,  128, 128, 8);
    k_conv_w<<<nb(4  * 3 * 256, BT), BT, 0, stream>>>(w_out,   wob,    128, CLS, 3);

    // degree counts (same for both layers)
    k_init_cnt<<<nb(NNODES, BT), BT, 0, stream>>>(cnt, NNODES);
    k_count<<<nb(NEDGES, BT), BT, 0, stream>>>(ei, cnt);

    const int edgeBlocks = nb((long)(NEDGES / 16) * 32, BT);
    const int nodeBlocks = nb((long)(NNODES / 16) * 32, BT);

    // ----- layer 1 -----
    k_init_aggr<<<nb((long)NNODES * F, BT), BT, 0, stream>>>(sum, mx, NNODES * F);
    k_edge<<<edgeBlocks, BT, 0, stream>>>(xb, wpre1, b1_pre, ei, sum, mx);
    k_post<<<nodeBlocks, BT, 0, stream>>>(xb, sum, mx, cnt, wpost1, b1_post,
                                          (unsigned short*)tb);
    k_lin<<<nodeBlocks, BT, 0, stream>>>(tb, wlin1, b1_lin, (unsigned short*)h1b);

    // ----- layer 2 -----
    k_init_aggr<<<nb((long)NNODES * F, BT), BT, 0, stream>>>(sum, mx, NNODES * F);
    k_edge<<<edgeBlocks, BT, 0, stream>>>(h1b, wpre2, b2_pre, ei, sum, mx);
    k_post<<<nodeBlocks, BT, 0, stream>>>(h1b, sum, mx, cnt, wpost2, b2_post,
                                          (unsigned short*)tb);
    k_lin<<<nodeBlocks, BT, 0, stream>>>(tb, wlin2, b2_lin, (unsigned short*)h2b);

    // ----- classifier -----
    k_out<<<nodeBlocks, BT, 0, stream>>>(h2b, wob, b_out, (float*)d_out);
}
